// PixtralAttention_86088324481840
// MI455X (gfx1250) — compile-verified, matrix operand order checked by reading, not run on hardware
//
#include <hip/hip_runtime.h>
#include <hip/hip_bf16.h>

// Problem constants (from reference): B=1, S=4096, H=16, D=64, E=1024
#define S_ 4096
#define H_ 16
#define D_ 64
#define E_ 1024
#define SCALE_ 0.125f   // D^-0.5

typedef __bf16 bf16;
typedef __bf16 v16bf __attribute__((ext_vector_type(16)));
typedef __bf16 v8bf  __attribute__((ext_vector_type(8)));
typedef float  v8f   __attribute__((ext_vector_type(8)));
typedef unsigned int v4u __attribute__((ext_vector_type(4)));
typedef int v4i __attribute__((ext_vector_type(4)));
typedef int v8i __attribute__((ext_vector_type(8)));

// ---------------------------------------------------------------------------
// 16x32 bf16 fragment load (A-layout per CDNA5 ISA 7.12.2):
//   lane L (0..31): row = L&15, K-base = (L>>4)*8
//   VGPRs 0-3 hold K = kb..kb+7 (contiguous, one b128), VGPRs 4-7 hold K+16.
// Works for A fragments (rows of A) and B fragments (rows of B^T / columns
// of B), from either global memory or LDS.
// ---------------------------------------------------------------------------
__device__ __forceinline__ v16bf load_frag(const bf16* base, int ld) {
  const int lane = threadIdx.x & 31;
  const int row  = lane & 15;
  const int kb   = (lane >> 4) << 3;
  const bf16* p  = base + (size_t)row * ld + kb;
  v8bf lo = *(const v8bf*)(p);
  v8bf hi = *(const v8bf*)(p + 16);
  v16bf r;
#pragma unroll
  for (int i = 0; i < 8; ++i) { r[i] = lo[i]; r[i + 8] = hi[i]; }
  return r;
}

__device__ __forceinline__ v8f wmma_bf16(v16bf a, v16bf b, v8f c) {
  return __builtin_amdgcn_wmma_f32_16x16x32_bf16(
      /*neg_a=*/false, a, /*neg_b=*/false, b,
      /*c_mod=*/(short)0, c, /*reuse_a=*/false, /*reuse_b=*/false);
}

// ---------------------------------------------------------------------------
// Tensor Data Mover: async-load a 64-row x 32-col bf16 tile (row stride `ld`
// elements) from global memory into LDS. D# built per CDNA5 ISA section 8:
//   group0: count=1 | lds_addr | global_addr[56:0] | type=2
//   group1: data_size=2B, tensor_dim0=32, tensor_dim1=64, tile_dim0=32,
//           tile_dim1=64, tensor_dim0_stride=ld  (2-D tile, no pad/iterate,
//           workgroup_mask=0 since we are not in a cluster)
//   groups 2/3: zero (<= 2-D tensor)
// Tracked by TENSORcnt; waited with s_wait_tensorcnt.
// ---------------------------------------------------------------------------
__device__ __forceinline__ void tdm_load_tile(bf16* lds_dst, const bf16* gsrc,
                                              int ld) {
  const unsigned int lds_addr = (unsigned int)(uintptr_t)lds_dst;
  const unsigned long long ga = (unsigned long long)(uintptr_t)gsrc;
  v4u g0;
  g0[0] = 1u;                                   // count=1 (valid descriptor)
  g0[1] = lds_addr;                             // LDS byte address
  g0[2] = (unsigned int)ga;                     // global_addr[31:0]
  g0[3] = (unsigned int)(ga >> 32) | (2u << 30);// global_addr[56:32] | type=2
  v8i g1;
  g1[0] = (int)(1u << 16);                      // data_size=1 (2 bytes)
  g1[1] = (int)(32u << 16);                     // tensor_dim0[15:0]=32
  g1[2] = (int)(64u << 16);                     // td0 hi=0 | tensor_dim1 lo=64
  g1[3] = (int)(32u << 16);                     // td1 hi=0 | tile_dim0=32
  g1[4] = 64;                                   // tile_dim1=64, tile_dim2=0
  g1[5] = ld;                                   // tensor_dim0_stride[31:0]
  g1[6] = 0;                                    // stride0 hi | stride1 lo
  g1[7] = 0;                                    // stride1 hi
  v4i gz = {0, 0, 0, 0};
#if defined(__clang_major__) && __clang_major__ >= 23
  v8i gz8 = {0, 0, 0, 0, 0, 0, 0, 0};
  __builtin_amdgcn_tensor_load_to_lds(g0, g1, gz, gz, gz8, 0);
#else
  __builtin_amdgcn_tensor_load_to_lds(g0, g1, gz, gz, 0);
#endif
}

// ---------------------------------------------------------------------------
// fp32 -> bf16 convert (hidden_states)
// ---------------------------------------------------------------------------
__global__ void cvt_bf16_kernel(const float* __restrict__ in,
                                bf16* __restrict__ out, int n) {
  int i = blockIdx.x * blockDim.x + threadIdx.x;
  if (i < n) out[i] = (bf16)in[i];
}

// ---------------------------------------------------------------------------
// LDS-tiled transpose + convert: out[c][r] = (bf16) in[r][c], in is RxC fp32.
// block (32,8), grid (C/32, R/32)
// ---------------------------------------------------------------------------
__global__ void transpose_w_kernel(const float* __restrict__ in,
                                   bf16* __restrict__ out, int R, int C) {
  __shared__ float t[32][33];
  const int bx = blockIdx.x * 32, by = blockIdx.y * 32;
#pragma unroll
  for (int i = threadIdx.y; i < 32; i += 8)
    t[i][threadIdx.x] = in[(size_t)(by + i) * C + bx + threadIdx.x];
  __syncthreads();
#pragma unroll
  for (int i = threadIdx.y; i < 32; i += 8)
    out[(size_t)(bx + i) * R + by + threadIdx.x] = (bf16)t[threadIdx.x][i];
}

// ---------------------------------------------------------------------------
// GEMM: C[MxN] = A[MxK] * B[KxN], B supplied transposed (Bt[N][K]).
// 128 threads = 4 waves, block tile 64x64, wave tile 32x32 (2x2 WMMA).
// TDM double-buffered LDS staging: wave 0 issues two TENSOR_LOAD_TO_LDS per
// 32-deep K-step (A 64x32, B 64x32). Pipeline: issue(next) ->
// s_wait_tensorcnt<=2 (per-wave in-order => current pair complete) ->
// barrier -> ds_load_b128 fragments + 4 WMMA -> barrier.
// OUT_F32: 0 -> store bf16, 1 -> store fp32.
// ---------------------------------------------------------------------------
template <int OUT_F32>
__global__ void gemm_wmma_kernel(const bf16* __restrict__ A,
                                 const bf16* __restrict__ Bt,
                                 void* __restrict__ C, int M, int N, int K) {
  __shared__ __align__(16) bf16 lA[2][64 * 32];
  __shared__ __align__(16) bf16 lB[2][64 * 32];
  const int wave = threadIdx.x >> 5;
  const int lane = threadIdx.x & 31;
  const int m0b = blockIdx.y * 64;
  const int n0b = blockIdx.x * 64;
  const int wm = (wave >> 1) * 32;
  const int wn = (wave & 1) * 32;
  const int nk = K >> 5;

  if (threadIdx.x < 32) {
    tdm_load_tile(&lA[0][0], A + (size_t)m0b * K, K);
    tdm_load_tile(&lB[0][0], Bt + (size_t)n0b * K, K);
  }

  v8f acc[2][2] = {};
  for (int kt = 0; kt < nk; ++kt) {
    const int cur = kt & 1;
    if (threadIdx.x < 32 && (kt + 1) < nk) {
      tdm_load_tile(&lA[cur ^ 1][0], A + (size_t)m0b * K + (kt + 1) * 32, K);
      tdm_load_tile(&lB[cur ^ 1][0], Bt + (size_t)n0b * K + (kt + 1) * 32, K);
    }
    if ((kt + 1) < nk)
      __builtin_amdgcn_s_wait_tensorcnt((short)2);  // current pair drained
    else
      __builtin_amdgcn_s_wait_tensorcnt((short)0);
    __syncthreads();  // current buffer visible to all waves

    const bf16* a = &lA[cur][0];
    const bf16* b = &lB[cur][0];
    v16bf a0 = load_frag(a + (wm + 0) * 32, 32);
    v16bf a1 = load_frag(a + (wm + 16) * 32, 32);
    v16bf b0 = load_frag(b + (wn + 0) * 32, 32);
    v16bf b1 = load_frag(b + (wn + 16) * 32, 32);
    acc[0][0] = wmma_bf16(a0, b0, acc[0][0]);
    acc[0][1] = wmma_bf16(a0, b1, acc[0][1]);
    acc[1][0] = wmma_bf16(a1, b0, acc[1][0]);
    acc[1][1] = wmma_bf16(a1, b1, acc[1][1]);

    __syncthreads();  // all reads consumed before next TDM overwrite
  }

  const int col = lane & 15, rsel = lane >> 4;
#pragma unroll
  for (int mi = 0; mi < 2; ++mi)
#pragma unroll
    for (int ni = 0; ni < 2; ++ni)
#pragma unroll
      for (int v = 0; v < 8; ++v) {
        const int r = m0b + wm + mi * 16 + v + 8 * rsel;
        const int c = n0b + wn + ni * 16 + col;
        if (OUT_F32)
          ((float*)C)[(size_t)r * N + c] = acc[mi][ni][v];
        else
          ((bf16*)C)[(size_t)r * N + c] = (bf16)acc[mi][ni][v];
      }
}

// ---------------------------------------------------------------------------
// RoPE on Q and K in place, bf16 [s][h*D+d]. One thread per (s,h,d<32) pair.
// out[d]    = x[d]*cos[d]       - x[d+32]*sin[d]
// out[d+32] = x[d+32]*cos[d+32] + x[d]   *sin[d+32]
// ---------------------------------------------------------------------------
__global__ void rope_kernel(bf16* __restrict__ Q, bf16* __restrict__ K,
                            const float* __restrict__ cosp,
                            const float* __restrict__ sinp) {
  int idx = blockIdx.x * blockDim.x + threadIdx.x;
  int d = idx & 31;
  int h = (idx >> 5) & (H_ - 1);
  int s = idx >> 9;
  if (s >= S_) return;
  const size_t base = (size_t)s * E_ + h * D_;
  const float c1 = cosp[s * D_ + d],      s1 = sinp[s * D_ + d];
  const float c2 = cosp[s * D_ + d + 32], s2 = sinp[s * D_ + d + 32];
  float q1 = (float)Q[base + d], q2 = (float)Q[base + d + 32];
  Q[base + d]      = (bf16)(q1 * c1 - q2 * s1);
  Q[base + d + 32] = (bf16)(q2 * c2 + q1 * s2);
  float k1 = (float)K[base + d], k2 = (float)K[base + d + 32];
  K[base + d]      = (bf16)(k1 * c1 - k2 * s1);
  K[base + d + 32] = (bf16)(k2 * c2 + k1 * s2);
}

// ---------------------------------------------------------------------------
// V [s][e] -> Vt [e][s] (== [h][d][s]) so PV B-fragments are contiguous in K=t.
// ---------------------------------------------------------------------------
__global__ void transpose_v_kernel(const bf16* __restrict__ V,
                                   bf16* __restrict__ Vt) {
  int idx = blockIdx.x * blockDim.x + threadIdx.x;
  int e = idx & (E_ - 1);
  int s = idx >> 10;
  if (s >= S_) return;
  Vt[(size_t)e * S_ + s] = V[(size_t)s * E_ + e];
}

// ---------------------------------------------------------------------------
// Flash attention. grid = (S/64, H), block = 128 (4 waves).
// Each wave owns 16 query rows; iterates over keys in chunks of 32.
// Per chunk: 4 WMMA (QK^T over d=64, two 16-col subtiles), online softmax via
// half-wave __shfl_xor reductions (C layout: row = VGPR, cols = 16 lanes),
// P re-laid-out C->A through a per-wave 1KB LDS bounce, 4 WMMA for PV.
// K/V/mask operands are L2-resident (192MB L2), streamed directly.
// ---------------------------------------------------------------------------
__global__ void flash_attn_kernel(const bf16* __restrict__ Q,
                                  const bf16* __restrict__ K,
                                  const bf16* __restrict__ Vt,
                                  const float* __restrict__ mask,
                                  bf16* __restrict__ O) {
  __shared__ __align__(16) bf16 plds[4][16 * 32];  // per-wave P tile
  const int wave = threadIdx.x >> 5;
  const int lane = threadIdx.x & 31;
  const int h = blockIdx.y;
  const int s0 = blockIdx.x * 64 + wave * 16;

  const bf16* Qh  = Q  + (size_t)s0 * E_ + h * D_;  // ld = E
  const bf16* Kh  = K  + (size_t)h * D_;            // row t at t*E
  const bf16* Vth = Vt + (size_t)h * D_ * S_;       // [d][t], ld = S

  const v16bf qf0 = load_frag(Qh, E_);       // d = 0..31
  const v16bf qf1 = load_frag(Qh + 32, E_);  // d = 32..63

  v8f acc[4] = {};
  float mrow[8], lrow[8];
#pragma unroll
  for (int v = 0; v < 8; ++v) { mrow[v] = -1e30f; lrow[v] = 0.0f; }

  const int col = lane & 15, hsel = lane >> 4;
  bf16* pl = plds[wave];

  for (int t0 = 0; t0 < S_; t0 += 32) {
    // ---- S = Q K^T for this 16x32 chunk (two 16x16 C tiles) ----
    v8f st[2] = {};
#pragma unroll
    for (int ti = 0; ti < 2; ++ti) {
      v16bf kb0 = load_frag(Kh + (size_t)(t0 + ti * 16) * E_, E_);
      v16bf kb1 = load_frag(Kh + (size_t)(t0 + ti * 16) * E_ + 32, E_);
      st[ti] = wmma_bf16(qf0, kb0, st[ti]);
      st[ti] = wmma_bf16(qf1, kb1, st[ti]);
    }

    // ---- scale + mask, online softmax ----
#pragma unroll
    for (int v = 0; v < 8; ++v) {
      const int r = s0 + v + 8 * hsel;
      st[0][v] = st[0][v] * SCALE_ + mask[(size_t)r * S_ + t0 + col];
      st[1][v] = st[1][v] * SCALE_ + mask[(size_t)r * S_ + t0 + 16 + col];

      float cmax = fmaxf(st[0][v], st[1][v]);
#pragma unroll
      for (int off = 8; off >= 1; off >>= 1)
        cmax = fmaxf(cmax, __shfl_xor(cmax, off, 32));
      const float newm = fmaxf(mrow[v], cmax);
      const float corr = __expf(mrow[v] - newm);
      mrow[v] = newm;

      const float p0 = __expf(st[0][v] - newm);
      const float p1 = __expf(st[1][v] - newm);
      st[0][v] = p0;
      st[1][v] = p1;
      float csum = p0 + p1;
#pragma unroll
      for (int off = 8; off >= 1; off >>= 1)
        csum += __shfl_xor(csum, off, 32);
      lrow[v] = lrow[v] * corr + csum;

      acc[0][v] *= corr; acc[1][v] *= corr;
      acc[2][v] *= corr; acc[3][v] *= corr;
    }

    // ---- P: C layout -> A layout via per-wave LDS bounce ----
#pragma unroll
    for (int v = 0; v < 8; ++v) {
      pl[(v + 8 * hsel) * 32 + col]      = (bf16)st[0][v];
      pl[(v + 8 * hsel) * 32 + 16 + col] = (bf16)st[1][v];
    }
    asm volatile("s_wait_dscnt 0x0" ::: "memory");
    const v16bf pf = load_frag(pl, 32);

    // ---- O += P @ V  (4 N-tiles of d) ----
#pragma unroll
    for (int j = 0; j < 4; ++j) {
      v16bf vf = load_frag(Vth + (size_t)(j * 16) * S_ + t0, S_);
      acc[j] = wmma_bf16(pf, vf, acc[j]);
    }
  }

  // ---- normalize and store bf16 [s][h*D+d] ----
#pragma unroll
  for (int j = 0; j < 4; ++j)
#pragma unroll
    for (int v = 0; v < 8; ++v) {
      const int r = s0 + v + 8 * hsel;
      const int c = h * D_ + j * 16 + col;
      O[(size_t)r * E_ + c] = (bf16)(acc[j][v] / lrow[v]);
    }
}

// ---------------------------------------------------------------------------
// Workspace layout (bytes), total 56 MiB (assumed <= ws_size):
//   hbf 0..8M | WqT 8M | WkT 10M | WvT 12M | WoT 14M |
//   Q 16M | K 24M | V 32M | Vt 40M | O 48M
// ---------------------------------------------------------------------------
extern "C" void kernel_launch(void* const* d_in, const int* in_sizes, int n_in,
                              void* d_out, int out_size, void* d_ws,
                              size_t ws_size, hipStream_t stream) {
  const float* hidden = (const float*)d_in[0];
  const float* cosp   = (const float*)d_in[1];
  const float* sinp   = (const float*)d_in[2];
  const float* mask   = (const float*)d_in[3];
  const float* Wq     = (const float*)d_in[4];
  const float* Wk     = (const float*)d_in[5];
  const float* Wv     = (const float*)d_in[6];
  const float* Wo     = (const float*)d_in[7];

  char* ws = (char*)d_ws;
  const size_t MB = 1024 * 1024;
  bf16* hbf = (bf16*)(ws + 0);
  bf16* WqT = (bf16*)(ws + 8 * MB);
  bf16* WkT = (bf16*)(ws + 10 * MB);
  bf16* WvT = (bf16*)(ws + 12 * MB);
  bf16* WoT = (bf16*)(ws + 14 * MB);
  bf16* Qb  = (bf16*)(ws + 16 * MB);
  bf16* Kb  = (bf16*)(ws + 24 * MB);
  bf16* Vb  = (bf16*)(ws + 32 * MB);
  bf16* Vt  = (bf16*)(ws + 40 * MB);
  bf16* Ob  = (bf16*)(ws + 48 * MB);

  // 1) convert activations
  {
    const int n = S_ * E_;
    cvt_bf16_kernel<<<(n + 255) / 256, 256, 0, stream>>>(hidden, hbf, n);
  }
  // 2) transpose+convert weights
  {
    dim3 g(E_ / 32, E_ / 32), b(32, 8);
    transpose_w_kernel<<<g, b, 0, stream>>>(Wq, WqT, E_, E_);
    transpose_w_kernel<<<g, b, 0, stream>>>(Wk, WkT, E_, E_);
    transpose_w_kernel<<<g, b, 0, stream>>>(Wv, WvT, E_, E_);
    transpose_w_kernel<<<g, b, 0, stream>>>(Wo, WoT, E_, E_);
  }
  // 3) Q/K/V projections (bf16 out, TDM-staged)
  {
    dim3 g(E_ / 64, S_ / 64);
    gemm_wmma_kernel<0><<<g, 128, 0, stream>>>(hbf, WqT, Qb, S_, E_, E_);
    gemm_wmma_kernel<0><<<g, 128, 0, stream>>>(hbf, WkT, Kb, S_, E_, E_);
    gemm_wmma_kernel<0><<<g, 128, 0, stream>>>(hbf, WvT, Vb, S_, E_, E_);
  }
  // 4) RoPE on Q,K
  {
    const int n = S_ * H_ * 32;
    rope_kernel<<<(n + 255) / 256, 256, 0, stream>>>(Qb, Kb, cosp, sinp);
  }
  // 5) V -> V^T
  {
    const int n = S_ * E_;
    transpose_v_kernel<<<(n + 255) / 256, 256, 0, stream>>>(Vb, Vt);
  }
  // 6) flash attention
  {
    dim3 g(S_ / 64, H_);
    flash_attn_kernel<<<g, 128, 0, stream>>>(Qb, Kb, Vt, mask, Ob);
  }
  // 7) output projection (fp32 out)
  {
    dim3 g(E_ / 64, S_ / 64);
    gemm_wmma_kernel<1><<<g, 128, 0, stream>>>(Ob, WoT, (float*)d_out, S_, E_,
                                               E_);
  }
  (void)in_sizes; (void)n_in; (void)out_size; (void)ws_size;
}
